// VectorQuantize_56392920596576
// MI455X (gfx1250) — matrix operand order repaired and checked by reading, not compiled, not run on hardware
//
#include <hip/hip_runtime.h>
#include <hip/hip_bf16.h>

// ---------------------------------------------------------------------------
// VQ: x [16,256,4096] f32, embed [1024,256] f32
// out = [quantized_st (16*256*4096) | indices-as-float (65536) | loss | util]
// ---------------------------------------------------------------------------

typedef __attribute__((ext_vector_type(16))) __bf16 v16bf;
typedef __attribute__((ext_vector_type(8)))  float  v8f;
typedef __attribute__((ext_vector_type(4)))  unsigned v4u;
typedef __attribute__((ext_vector_type(8)))  unsigned v8u;

#define DIM       256
#define TT        4096
#define BB        16
#define NTOK      65536          // BB*TT
#define KCODES    1024
#define QOFF      16777216       // BB*DIM*TT
#define LDSPITCH  264            // 256 + 8 bf16 pad = 528B row pitch (TDM-padded)

__device__ __forceinline__ unsigned short f32_to_bf16_rne(float f) {
    unsigned u = __float_as_uint(f);
    unsigned r = u + 0x7FFFu + ((u >> 16) & 1u);
    return (unsigned short)(r >> 16);
}

// ---------------------------------------------------------------------------
// TDM: DMA one 16x256 bf16 tile (one code tile of embed) into LDS.
// D# pad feature inserts 16B after every 512B row -> LDS pitch 528B = LDSPITCH.
// 2D form: descriptor groups 0 (4 SGPRs) + 1 (8 SGPRs), VADDR2/3 = NULL.
// ---------------------------------------------------------------------------
__device__ __forceinline__ void tdm_load_tile(const unsigned short* gsrc,
                                              unsigned lds_off) {
    unsigned long long ga = (unsigned long long)(uintptr_t)gsrc;
    v4u g0;
    g0[0] = 1u;                                   // count=1; user-mode flags 0
    g0[1] = lds_off;                              // lds_addr (bytes)
    g0[2] = (unsigned)(ga & 0xFFFFFFFFull);       // global_addr[31:0]
    g0[3] = (unsigned)((ga >> 32) & 0x1FFFFFFull) // global_addr[56:32]
          | (2u << 30);                           // type = 2 ("image")
    v8u g1;
    g1[0] = (1u << 16)      // data_size = 1 (2 bytes, bf16)
          | (1u << 20)      // pad_enable
          | (6u << 22)      // pad_interval: 2^(6+1)=128 dwords = 512B (one row)
          | (3u << 25);     // pad_amount: 3+1 = 4 dwords = 16B
    g1[1] = (256u  << 16);  // tensor_dim0 = 256 (row length, elements)
    g1[2] = (1024u << 16);  // tensor_dim0 hi=0 | tensor_dim1 = 1024 (lo16)
    g1[3] = (256u  << 16);  // tensor_dim1 hi=0 | tile_dim0 = 256
    g1[4] = 16u;            // tile_dim1 = 16 rows | tile_dim2 = 0
    g1[5] = 256u;           // tensor_dim0_stride[31:0] = 256 elements
    g1[6] = 0u;             // stride0 hi | tensor_dim1_stride lo
    g1[7] = 0u;
    asm volatile("tensor_load_to_lds %0, %1" :: "s"(g0), "s"(g1) : "memory");
}

// ---------------------------------------------------------------------------
// Kernel 0: zero counts
// ---------------------------------------------------------------------------
__global__ __launch_bounds__(1024) void vq_zero(int* __restrict__ counts) {
    counts[threadIdx.x] = 0;
}

// ---------------------------------------------------------------------------
// Kernel 1: embed f32 -> bf16 hi/lo split + exact fp32 e_sq
// grid = 1024 blocks (one code row), 256 threads
// ---------------------------------------------------------------------------
__global__ __launch_bounds__(256) void vq_prep_embed(
    const float* __restrict__ embed,
    unsigned short* __restrict__ ehi, unsigned short* __restrict__ elo,
    float* __restrict__ esq) {
    __shared__ float ssum[8];
    const int k = blockIdx.x, d = threadIdx.x;
    const int idx = (k << 8) + d;
    float v = embed[idx];
    unsigned short hb = f32_to_bf16_rne(v);
    float hf = __uint_as_float((unsigned)hb << 16);
    ehi[idx] = hb;
    elo[idx] = f32_to_bf16_rne(v - hf);
    float sq = v * v;
    #pragma unroll
    for (int off = 16; off >= 1; off >>= 1) sq += __shfl_xor(sq, off);
    if ((threadIdx.x & 31) == 0) ssum[threadIdx.x >> 5] = sq;
    __syncthreads();
    if (threadIdx.x == 0) {
        float t = 0.f;
        #pragma unroll
        for (int w = 0; w < 8; ++w) t += ssum[w];
        esq[k] = t;
    }
}

// ---------------------------------------------------------------------------
// Kernel 2: WMMA argmin. 256 thr = 8 waves; wave = one 16-token tile.
// A (x tile, bf16 hi/lo) held in registers; embed tiles double-buffered in
// LDS via TDM (wave 0 issues, overlapped with WMMA on the other buffer).
// ---------------------------------------------------------------------------
__global__ __launch_bounds__(256) void vq_argmin(
    const float* __restrict__ x,
    const unsigned short* __restrict__ ehi, const unsigned short* __restrict__ elo,
    const float* __restrict__ esq,
    int* __restrict__ idxw, int* __restrict__ counts,
    float* __restrict__ d2arr, float* __restrict__ out_idx) {

    __shared__ unsigned short shh[2][16 * LDSPITCH];
    __shared__ unsigned short shl[2][16 * LDSPITCH];

    const int tid   = threadIdx.x;
    const int wave  = tid >> 5;
    const int lane  = tid & 31;
    const int tile  = blockIdx.x * 8 + wave;
    const int n0    = tile << 4;
    const int bb    = n0 >> 12;          // batch
    const int t0    = n0 & 4095;         // time offset
    const int m     = lane & 15;         // A row (token) / B column (code)
    const int khalf = lane >> 4;         // A K-half select
    const int kb    = khalf << 4;        // B K-base (0 or 16)

    // Prologue: kick off DMA of code tile 0 while we build A in registers.
    if (wave == 0) {
        tdm_load_tile(ehi, (unsigned)(uintptr_t)(&shh[0][0]));
        tdm_load_tile(elo, (unsigned)(uintptr_t)(&shl[0][0]));
    }

    // ---- Load A: 16 tokens x 256 d, split f32 -> bf16 hi/lo, WMMA layout ----
    v16bf ah[8], al[8];
    float xsp = 0.f;
    const float* xb = x + ((size_t)bb << 20);
    #pragma unroll
    for (int s = 0; s < 8; ++s) {
        union { unsigned short us[16]; v16bf v; } H, L;
        #pragma unroll
        for (int e = 0; e < 16; ++e) {
            const int kk = e + khalf * 16;
            const int d  = (s << 5) + kk;
            float v = xb[((size_t)d << 12) + (size_t)(t0 + m)];
            unsigned short hb = f32_to_bf16_rne(v);
            float hf = __uint_as_float((unsigned)hb << 16);
            H.us[e] = hb;
            L.us[e] = f32_to_bf16_rne(v - hf);
            xsp += v * v;
        }
        ah[s] = H.v; al[s] = L.v;
    }
    const float xsqtok = xsp + __shfl_xor(xsp, 16);   // ||x_m||^2 (lanes m,m+16)

    float best[8]; int bidx[8];
    #pragma unroll
    for (int r = 0; r < 8; ++r) { best[r] = 3.4e38f; bidx[r] = 0; }

    // ---- Stream 64 code tiles through double-buffered LDS via TDM ----
    for (int ct = 0; ct < 64; ++ct) {
        const int buf = ct & 1;
        if (wave == 0) __builtin_amdgcn_s_wait_tensorcnt(0);  // tile ct landed
        __syncthreads();  // all may read buf; all done reading buf^1
        if (wave == 0 && ct + 1 < 64) {   // DMA next tile, overlapped w/ WMMA
            const size_t goff = ((size_t)(ct + 1) << 4) << 8;   // (ct+1)*16*256
            tdm_load_tile(ehi + goff, (unsigned)(uintptr_t)(&shh[buf ^ 1][0]));
            tdm_load_tile(elo + goff, (unsigned)(uintptr_t)(&shl[buf ^ 1][0]));
        }

        const float eq = esq[(ct << 4) + m];
        v8f acc = {0.f, 0.f, 0.f, 0.f, 0.f, 0.f, 0.f, 0.f};

        #pragma unroll
        for (int s = 0; s < 8; ++s) {
            union { uint4 q[2]; v16bf v; } BH, BL;
            const uint4* ph = (const uint4*)(&shh[buf][m * LDSPITCH + (s << 5) + kb]);
            const uint4* pl = (const uint4*)(&shl[buf][m * LDSPITCH + (s << 5) + kb]);
            BH.q[0] = ph[0]; BH.q[1] = ph[1];
            BL.q[0] = pl[0]; BL.q[1] = pl[1];
            // fp32-fidelity dot via 3-term bf16 split: hh + hl + lh
            acc = __builtin_amdgcn_wmma_f32_16x16x32_bf16(false, ah[s], false, BH.v,
                                                          (short)0, acc, false, false);
            acc = __builtin_amdgcn_wmma_f32_16x16x32_bf16(false, ah[s], false, BL.v,
                                                          (short)0, acc, false, false);
            acc = __builtin_amdgcn_wmma_f32_16x16x32_bf16(false, al[s], false, BH.v,
                                                          (short)0, acc, false, false);
        }

        const int code = (ct << 4) + m;
        #pragma unroll
        for (int r = 0; r < 8; ++r) {
            const float sc = eq - 2.0f * acc[r];   // d2 - ||x||^2 (const per token)
            if (sc < best[r]) { best[r] = sc; bidx[r] = code; }
        }
    }

    // ---- cross-lane lex-min: 16 lanes share one token row ----
    #pragma unroll
    for (int off = 8; off >= 1; off >>= 1) {
        #pragma unroll
        for (int r = 0; r < 8; ++r) {
            const float os = __shfl_xor(best[r], off);
            const int   oi = __shfl_xor(bidx[r], off);
            if (os < best[r] || (os == best[r] && oi < bidx[r])) {
                best[r] = os; bidx[r] = oi;
            }
        }
    }

    // gather ||x||^2 for the 8 tokens this lane-group owns
    float xs[8];
    #pragma unroll
    for (int r = 0; r < 8; ++r)
        xs[r] = __shfl(xsqtok, ((lane & 16) ? 8 : 0) + r);

    if ((lane & 15) == 0) {
        const int base = n0 + ((lane >> 4) << 3);
        #pragma unroll
        for (int r = 0; r < 8; ++r) {
            const int code = bidx[r];
            const float d2 = fmaxf(best[r] + xs[r], 0.0f);
            idxw[base + r]    = code;
            out_idx[base + r] = (float)code;
            d2arr[base + r]   = d2;
            atomicAdd(&counts[code], 1);
        }
    }
}

// ---------------------------------------------------------------------------
// Kernel 3: quantized_st[b,d,t] = embed[idx[n], d]  (exact fp32 gather)
// grid = 256 blocks (b x t-chunk), 256 threads = 256 consecutive tokens
// ---------------------------------------------------------------------------
__global__ __launch_bounds__(256) void vq_writeq(
    const float* __restrict__ embed, const int* __restrict__ idxw,
    float* __restrict__ outq) {
    const int bblk = blockIdx.x >> 4;
    const int t    = ((blockIdx.x & 15) << 8) + threadIdx.x;
    const int n    = (bblk << 12) + t;
    const float* er = embed + ((size_t)idxw[n] << 8);
    float* ob = outq + (((size_t)bblk) << 20) + t;
    #pragma unroll 4
    for (int d = 0; d < 256; ++d)
        ob[(size_t)d << 12] = er[d];   // lanes write consecutive t: coalesced
}

// ---------------------------------------------------------------------------
// Kernel 4: deterministic reductions -> loss, utilization
// ---------------------------------------------------------------------------
__global__ __launch_bounds__(1024) void vq_finalize(
    const float* __restrict__ d2arr, const int* __restrict__ counts,
    float* __restrict__ out_scalars) {
    __shared__ float sl[32];
    __shared__ int   su[32];
    const int tid = threadIdx.x;
    float s = 0.f;
    for (int k = tid; k < NTOK; k += 1024) s += d2arr[k];   // fixed order
    int used = (counts[tid] > 0) ? 1 : 0;
    #pragma unroll
    for (int off = 16; off >= 1; off >>= 1) {
        s    += __shfl_xor(s, off);
        used += __shfl_xor(used, off);
    }
    if ((tid & 31) == 0) { sl[tid >> 5] = s; su[tid >> 5] = used; }
    __syncthreads();
    if (tid == 0) {
        float ts = 0.f; int tu = 0;
        #pragma unroll
        for (int w = 0; w < 32; ++w) { ts += sl[w]; tu += su[w]; }
        out_scalars[0] = ts / 16777216.0f;        // mean over N*D
        out_scalars[1] = (float)tu / 1024.0f;     // codebook utilization
    }
}

// ---------------------------------------------------------------------------
extern "C" void kernel_launch(void* const* d_in, const int* in_sizes, int n_in,
                              void* d_out, int out_size, void* d_ws, size_t ws_size,
                              hipStream_t stream) {
    const float* x     = (const float*)d_in[0];
    const float* embed = (const float*)d_in[1];
    float* out = (float*)d_out;

    char* w = (char*)d_ws;
    unsigned short* ehi   = (unsigned short*)(w);            // 512 KB
    unsigned short* elo   = (unsigned short*)(w + 524288);   // 512 KB
    float*          esq   = (float*)(w + 1048576);           // 4 KB
    int*            idxw  = (int*)  (w + 1052672);           // 256 KB
    int*            counts= (int*)  (w + 1314816);           // 4 KB
    float*          d2arr = (float*)(w + 1318912);           // 256 KB

    vq_zero<<<1, 1024, 0, stream>>>(counts);
    vq_prep_embed<<<KCODES, 256, 0, stream>>>(embed, ehi, elo, esq);
    vq_argmin<<<NTOK / 16 / 8, 256, 0, stream>>>(x, ehi, elo, esq, idxw, counts,
                                                 d2arr, out + QOFF);
    vq_writeq<<<256, 256, 0, stream>>>(embed, idxw, out);
    vq_finalize<<<1, 1024, 0, stream>>>(d2arr, counts, out + QOFF + NTOK);
}
